// NodeModel_49830210568748
// MI455X (gfx1250) — compile-verified, matrix operand order checked by reading, not run on hardware
//
#include <hip/hip_runtime.h>

// ---------------------------------------------------------------------------
// Types for CDNA5 WMMA (wave32)
// ---------------------------------------------------------------------------
typedef __attribute__((ext_vector_type(16))) __bf16        v16bf;
typedef __attribute__((ext_vector_type(8)))  float         v8f;
typedef __attribute__((ext_vector_type(4)))  float         v4f;
typedef __attribute__((ext_vector_type(4)))  unsigned int  v4u;

union Frag {           // 32 bytes = 16 bf16 = one WMMA A/B fragment per lane
    v16bf bf;
    v4u   u[2];
};

#define N_NODES_DIM 128   // NODE_IN
#define EDGE_DIM    64    // EDGE_OUT
#define GLOB_DIM    64    // GLOB_IN
#define IN_DIM      256   // 128+64+64
#define HID_DIM     256
#define OUT_DIM     128
#define SH_STRIDE   272   // 256 + 16 bf16 pad -> 8-dword bank rotation

// ---------------------------------------------------------------------------
// Kernel 0: zero agg_sum [N*64] and cnt [N] (contiguous N*65 floats)
// ---------------------------------------------------------------------------
__global__ void zero_kernel(float* __restrict__ p, int n) {
    int i = blockIdx.x * blockDim.x + threadIdx.x;
    if (i < n) p[i] = 0.0f;
}

// ---------------------------------------------------------------------------
// Kernel 1: transpose + bf16-convert weights.
//   W1t[n][k] = bf16(W1[k][n])  (256 x 256)
//   W2t[n][k] = bf16(W2[k][n])  (128 x 256)
// ---------------------------------------------------------------------------
__global__ void weights_kernel(const float* __restrict__ W1,
                               const float* __restrict__ W2,
                               __bf16* __restrict__ W1t,
                               __bf16* __restrict__ W2t) {
    int idx = blockIdx.x * blockDim.x + threadIdx.x;
    if (idx < 256 * 256) {
        int n = idx & 255, k = idx >> 8;
        W1t[n * 256 + k] = (__bf16)W1[k * 256 + n];
    } else {
        int j = idx - 256 * 256;
        if (j < 128 * 256) {
            int n = j & 127, k = j >> 7;
            W2t[n * 256 + k] = (__bf16)W2[k * 128 + n];
        }
    }
}

// ---------------------------------------------------------------------------
// Kernel 2: scatter-add edges onto source nodes.
// 16 threads per edge, each handles 4 consecutive floats (float4 read).
// ---------------------------------------------------------------------------
__global__ void scatter_kernel(const float* __restrict__ edge_attr,
                               const int*   __restrict__ src,
                               float* __restrict__ agg_sum,
                               float* __restrict__ cnt,
                               int E) {
    long long t = (long long)blockIdx.x * blockDim.x + threadIdx.x;
    long long e = t >> 4;
    int chunk   = (int)(t & 15);
    if (e >= E) return;
    int s = src[e];
    v4f v = *(const v4f*)(edge_attr + e * 64 + chunk * 4);
    float* dst = agg_sum + (size_t)s * 64 + chunk * 4;
    __hip_atomic_fetch_add(dst + 0, v.x, __ATOMIC_RELAXED, __HIP_MEMORY_SCOPE_AGENT);
    __hip_atomic_fetch_add(dst + 1, v.y, __ATOMIC_RELAXED, __HIP_MEMORY_SCOPE_AGENT);
    __hip_atomic_fetch_add(dst + 2, v.z, __ATOMIC_RELAXED, __HIP_MEMORY_SCOPE_AGENT);
    __hip_atomic_fetch_add(dst + 3, v.w, __ATOMIC_RELAXED, __HIP_MEMORY_SCOPE_AGENT);
    if (chunk == 0)
        __hip_atomic_fetch_add(&cnt[s], 1.0f, __ATOMIC_RELAXED, __HIP_MEMORY_SCOPE_AGENT);
}

// ---------------------------------------------------------------------------
// Kernel 3: fused MLP. Block = 128 threads = 4 waves, 64-row tile.
// Wave w owns rows [tile+16w, tile+16w+16).
//   GEMM1: A built in registers from x/agg/u (bf16), B = W1t -> h (relu, bf16)
//          stored to LDS.
//   GEMM2: A = h from LDS, B = W2t -> out (f32, +b2).
// ---------------------------------------------------------------------------
__global__ __launch_bounds__(128) void mlp_kernel(
    const float* __restrict__ x,
    const float* __restrict__ agg_sum,
    const float* __restrict__ cnt,
    const float* __restrict__ u,
    const int*   __restrict__ batch,
    const __bf16* __restrict__ W1t,
    const float* __restrict__ b1,
    const __bf16* __restrict__ W2t,
    const float* __restrict__ b2,
    float* __restrict__ out,
    int N) {
    __shared__ __bf16 sH[64 * SH_STRIDE];   // 34,816 B

    const int tid  = threadIdx.x;
    const int lane = tid & 31;
    const int wv   = tid >> 5;          // wave id 0..3
    const int r    = lane & 15;         // row / col within 16-tile
    const int half = lane >> 4;         // 0 or 1
    const int rowBase = blockIdx.x * 64 + wv * 16;
    const int gr   = rowBase + r;       // this lane's global row
    const bool inb = gr < N;

    // Per-row scalars
    float inv  = 1.0f;
    int   bidx = 0;
    if (inb) {
        inv  = 1.0f / fmaxf(cnt[gr], 1.0f);
        bidx = batch[gr];
    }
    const float* xr = x       + (size_t)gr   * N_NODES_DIM;
    const float* ar = agg_sum + (size_t)gr   * EDGE_DIM;
    const float* ur = u       + (size_t)bidx * GLOB_DIM;

    // ---- Build A fragments (16x32 bf16 per K-step) directly in registers ----
    v16bf A[8];
#pragma unroll
    for (int kk = 0; kk < 8; ++kk) {
        const int kbase = kk * 32;                       // compile-time constant
        const float* src; float sc;
        if (kbase < N_NODES_DIM)                { src = xr + kbase;                     sc = 1.0f; }
        else if (kbase < N_NODES_DIM + EDGE_DIM){ src = ar + (kbase - N_NODES_DIM);     sc = inv;  }
        else                                    { src = ur + (kbase - N_NODES_DIM - EDGE_DIM); sc = 1.0f; }
        float f[16];
        if (inb) {
            const float* p0 = src + half * 8;            // K = k0 .. k0+7
            v4f q0 = *(const v4f*)(p0);
            v4f q1 = *(const v4f*)(p0 + 4);
            const float* p1 = p0 + 16;                   // K = k0+16 .. k0+23
            v4f q2 = *(const v4f*)(p1);
            v4f q3 = *(const v4f*)(p1 + 4);
            f[0]=q0.x*sc; f[1]=q0.y*sc; f[2]=q0.z*sc; f[3]=q0.w*sc;
            f[4]=q1.x*sc; f[5]=q1.y*sc; f[6]=q1.z*sc; f[7]=q1.w*sc;
            f[8]=q2.x*sc; f[9]=q2.y*sc; f[10]=q2.z*sc; f[11]=q2.w*sc;
            f[12]=q3.x*sc; f[13]=q3.y*sc; f[14]=q3.z*sc; f[15]=q3.w*sc;
        } else {
#pragma unroll
            for (int i = 0; i < 16; ++i) f[i] = 0.0f;
        }
        v16bf a;
#pragma unroll
        for (int i = 0; i < 16; ++i) a[i] = (__bf16)f[i];
        A[kk] = a;
    }

    // ---- GEMM1: 16 N-tiles of HIDDEN, K = 256 (8 WMMA steps each) ----
#pragma unroll
    for (int n = 0; n < 16; ++n) {
        v8f acc = {0.f, 0.f, 0.f, 0.f, 0.f, 0.f, 0.f, 0.f};
        const __bf16* wrow = W1t + (size_t)(n * 16 + r) * IN_DIM;
#pragma unroll
        for (int kk = 0; kk < 8; ++kk) {
            Frag b;
            const __bf16* bp = wrow + kk * 32 + half * 8;
            b.u[0] = *(const v4u*)(bp);
            b.u[1] = *(const v4u*)(bp + 16);
            acc = __builtin_amdgcn_wmma_f32_16x16x32_bf16(
                false, A[kk], false, b.bf, (short)0, acc, false, false);
        }
        const float bias = b1[n * 16 + r];
#pragma unroll
        for (int v = 0; v < 8; ++v) {
            float hv = fmaxf(acc[v] + bias, 0.0f);                    // relu
            sH[(wv * 16 + v + half * 8) * SH_STRIDE + n * 16 + r] = (__bf16)hv;
        }
    }

    __syncthreads();

    // ---- Load h fragments from LDS ----
    v16bf A2[8];
#pragma unroll
    for (int kk = 0; kk < 8; ++kk) {
        Frag a;
        const int off = (wv * 16 + r) * SH_STRIDE + kk * 32 + half * 8;
        a.u[0] = *(const v4u*)(&sH[off]);
        a.u[1] = *(const v4u*)(&sH[off + 16]);
        A2[kk] = a.bf;
    }

    // ---- GEMM2: 8 N-tiles of OUT_DIM, K = 256 ----
#pragma unroll
    for (int n = 0; n < 8; ++n) {
        v8f acc = {0.f, 0.f, 0.f, 0.f, 0.f, 0.f, 0.f, 0.f};
        const __bf16* wrow = W2t + (size_t)(n * 16 + r) * HID_DIM;
#pragma unroll
        for (int kk = 0; kk < 8; ++kk) {
            Frag b;
            const __bf16* bp = wrow + kk * 32 + half * 8;
            b.u[0] = *(const v4u*)(bp);
            b.u[1] = *(const v4u*)(bp + 16);
            acc = __builtin_amdgcn_wmma_f32_16x16x32_bf16(
                false, A2[kk], false, b.bf, (short)0, acc, false, false);
        }
        const float bias = b2[n * 16 + r];
#pragma unroll
        for (int v = 0; v < 8; ++v) {
            int grow = rowBase + v + half * 8;
            if (grow < N)
                out[(size_t)grow * OUT_DIM + n * 16 + r] = acc[v] + bias;
        }
    }
}

// ---------------------------------------------------------------------------
// Host launch
// ---------------------------------------------------------------------------
extern "C" void kernel_launch(void* const* d_in, const int* in_sizes, int n_in,
                              void* d_out, int out_size, void* d_ws, size_t ws_size,
                              hipStream_t stream) {
    const float* x         = (const float*)d_in[0];
    const int*   edge_idx  = (const int*)  d_in[1];   // [2, E]; row 0 = src
    const float* edge_attr = (const float*)d_in[2];
    const float* u         = (const float*)d_in[3];
    const int*   batch     = (const int*)  d_in[4];
    const float* W1        = (const float*)d_in[5];
    const float* b1        = (const float*)d_in[6];
    const float* W2        = (const float*)d_in[7];
    const float* b2        = (const float*)d_in[8];
    float*       out       = (float*)d_out;

    const int N = in_sizes[0] / N_NODES_DIM;   // 100000
    const int E = in_sizes[1] / 2;             // 1200000

    // Workspace layout (bytes): agg_sum | cnt | W1t | W2t  (all 16B aligned)
    char*   ws      = (char*)d_ws;
    float*  agg_sum = (float*)ws;                                    // N*64 f32
    float*  cnt     = agg_sum + (size_t)N * 64;                      // N f32
    __bf16* W1t     = (__bf16*)(ws + (size_t)N * 65 * sizeof(float));
    __bf16* W2t     = W1t + 256 * 256;

    // 0) zero accumulators (must happen every call: graph replays)
    {
        int n = N * 65;
        zero_kernel<<<(n + 255) / 256, 256, 0, stream>>>(agg_sum, n);
    }
    // 1) bf16-transposed weights
    weights_kernel<<<(256 * 256 + 128 * 256 + 255) / 256, 256, 0, stream>>>(
        W1, W2, W1t, W2t);
    // 2) edge scatter (atomic f32 adds)
    {
        long long threads = (long long)E * 16;
        int blocks = (int)((threads + 255) / 256);
        scatter_kernel<<<blocks, 256, 0, stream>>>(edge_attr, edge_idx,
                                                   agg_sum, cnt, E);
    }
    // 3) fused gather + 2-layer MLP with bf16 WMMA
    mlp_kernel<<<(N + 63) / 64, 128, 0, stream>>>(
        x, agg_sum, cnt, u, batch, W1t, b1, W2t, b2, out, N);
}
